// AGCLSTM_35296041239235
// MI455X (gfx1250) — compile-verified, matrix operand order checked by reading, not run on hardware
//
#include <hip/hip_runtime.h>

#define N_NODES 256
#define HID     128
#define TSEQ    16
#define BATCH   32
#define ALPHA   0.01f

#define XLD 136   // padded row stride for 128-wide LDS rows (bf16)
#define HLD 264   // padded row stride for 256-wide LDS rows (bf16)

typedef __bf16 bf16;
typedef bf16  v16bf __attribute__((ext_vector_type(16)));
typedef bf16  v8bf  __attribute__((ext_vector_type(8)));
typedef bf16  v4bf  __attribute__((ext_vector_type(4)));
typedef bf16  v2bf  __attribute__((ext_vector_type(2)));
typedef float v8f   __attribute__((ext_vector_type(8)));
typedef float v4f   __attribute__((ext_vector_type(4)));

struct GatArgs { const float* W[8]; const float* a[8]; };

// ---------------- WMMA fragment helpers (wave32, 16x16x32 bf16) ----------------
// Per-lane data = two contiguous 16B chunks: K in {kb..kb+7} and {kb+16..kb+23},
// kb = (lane>=16)*8, "row" = lane&15.  Works for A tiles (row = M) and for
// K-major (transposed) B tiles (row = N).  ld*2 and tile bases must be 16B-aligned.
__device__ __forceinline__ v16bf load_frag_kmaj(const bf16* p, int ld) {
  const int lane = threadIdx.x & 31;
  const bf16* q  = p + (lane & 15) * ld + ((lane >> 4) << 3);
  v8bf lo = *(const v8bf*)q;
  v8bf hi = *(const v8bf*)(q + 16);
  return __builtin_shufflevector(lo, hi, 0, 1, 2, 3, 4, 5, 6, 7,
                                 8, 9, 10, 11, 12, 13, 14, 15);
}

// Store a 16x16 f32 D-tile transposed as bf16: element (m,n) -> base[n*ld + m].
// Per lane the 8 values are contiguous -> single ds_store_b128.
__device__ __forceinline__ void store_tile_bf16_T(bf16* base, int ld, v8f acc) {
  const int lane = threadIdx.x & 31;
  const int col = lane & 15, hi = lane >> 4;
  v8bf pk;
#pragma unroll
  for (int v = 0; v < 8; ++v) pk[v] = (bf16)acc[v];
  *(v8bf*)(base + col * ld + hi * 8) = pk;
}

__device__ __forceinline__ v8f wmma_bf16(v16bf a, v16bf b, v8f c) {
  return __builtin_amdgcn_wmma_f32_16x16x32_bf16(false, a, false, b, (short)0, c,
                                                 false, false);
}

__device__ __forceinline__ float sigmf(float x) { return 1.0f / (1.0f + __expf(-x)); }

// ---------------- GAT kernel: one (batch, unit) 256x128 block per WG ----------------
// unit u = gate*2 + src (src 0 = x_t, 1 = h).  gout layout: [u][b][j][n] (j-major).
__global__ __launch_bounds__(256) void gat_kernel(const float* __restrict__ x,
                                                  const float* __restrict__ hbuf,
                                                  GatArgs args,
                                                  float* __restrict__ gout, int t) {
  extern __shared__ char smem[];
  bf16*  sWt = (bf16*)smem;                  // W^T  : [j][k], 128 x XLD
  bf16*  sX  = sWt + HID * XLD;              // X    : [n][k], 256 x XLD
  bf16*  sHt = sX + N_NODES * XLD;           // h^T  : [j][n], 128 x HLD
  float* f1  = (float*)(sHt + HID * HLD);
  float* f2  = f1 + N_NODES;
  float* rmx = f2 + N_NODES;
  float* rsi = rmx + N_NODES;

  const int b = blockIdx.x, u = blockIdx.y;
  const float* __restrict__ W  = args.W[u];
  const float* __restrict__ av = args.a[u];
  const int tid = threadIdx.x;

  // W^T into LDS (coalesced global reads, packed b32 LDS stores)
  for (int i = tid; i < HID * HID / 2; i += 256) {
    const int j = i & 127, k = (i >> 7) * 2;
    v2bf pk = {(bf16)W[k * HID + j], (bf16)W[(k + 1) * HID + j]};
    *(v2bf*)(sWt + j * XLD + k) = pk;
  }
  if ((u & 1) == 0) {
    // x_t : row-major (n,k) -> float4 reads, v4bf stores; prefetch next slice to L2
    for (int i = tid; i < N_NODES * HID / 4; i += 256) {
      const int n = i >> 5, c4 = (i & 31) << 2;
      const float* xp = x + ((size_t)(b * N_NODES + n) * TSEQ + t) * HID + c4;
      const v4f xv = *(const v4f*)xp;
      if (t + 1 < TSEQ) __builtin_prefetch(xp + HID, 0, 1);   // global_prefetch
      v4bf pk = {(bf16)xv.x, (bf16)xv.y, (bf16)xv.z, (bf16)xv.w};
      *(v4bf*)(sX + n * XLD + c4) = pk;
    }
  } else {
    // h : stored (b,j,n) j-major -> coalesced reads, paired b32 LDS stores
    const float* hp = hbuf + (size_t)b * HID * N_NODES;
    for (int i = tid; i < N_NODES * HID / 2; i += 256) {
      const int n = i & 255, j = (i >> 8) * 2;
      v2bf pk = {(bf16)hp[j * N_NODES + n], (bf16)hp[(j + 1) * N_NODES + n]};
      *(v2bf*)(sX + n * XLD + j) = pk;
    }
  }
  __syncthreads();

  // ---- h = X @ W   (M=256, N=128, K=128); wave owns 2 M-tiles, B-frags shared ----
  const int wv = tid >> 5, lane = tid & 31, col = lane & 15, hi = lane >> 4;
  const int mt0 = wv * 2, mt1 = wv * 2 + 1;
  {
    v16bf a0[4], a1[4];
#pragma unroll
    for (int ks = 0; ks < 4; ++ks) {
      a0[ks] = load_frag_kmaj(sX + mt0 * 16 * XLD + ks * 32, XLD);
      a1[ks] = load_frag_kmaj(sX + mt1 * 16 * XLD + ks * 32, XLD);
    }
#pragma unroll
    for (int nt = 0; nt < 8; ++nt) {
      v8f c0 = {0.f, 0.f, 0.f, 0.f, 0.f, 0.f, 0.f, 0.f};
      v8f c1 = {0.f, 0.f, 0.f, 0.f, 0.f, 0.f, 0.f, 0.f};
#pragma unroll
      for (int ks = 0; ks < 4; ++ks) {
        const v16bf bfr = load_frag_kmaj(sWt + nt * 16 * XLD + ks * 32, XLD);
        c0 = wmma_bf16(a0[ks], bfr, c0);
        c1 = wmma_bf16(a1[ks], bfr, c1);
      }
      store_tile_bf16_T(sHt + nt * 16 * HLD + mt0 * 16, HLD, c0);
      store_tile_bf16_T(sHt + nt * 16 * HLD + mt1 * 16, HLD, c1);
    }
  }
  __syncthreads();

  // ---- f1 = h@a1, f2 = h@a2 (one node per thread; h^T reads are lane-contiguous) ----
  {
    float s1 = 0.f, s2 = 0.f;
    for (int j = 0; j < HID; ++j) {
      const float hv = (float)sHt[j * HLD + tid];
      s1 += hv * av[j];
      s2 += hv * av[HID + j];
    }
    f1[tid] = s1; f2[tid] = s2;
  }
  __syncthreads();

  // ---- softmax row stats over on-the-fly e[i,j] = lrelu(f1[i]+f2[j]) ----
  {
    const float fi = f1[tid];
    float m = -1e30f;
    for (int j = 0; j < N_NODES; ++j) {
      float e = fi + f2[j]; e = e > 0.f ? e : ALPHA * e;
      m = fmaxf(m, e);
    }
    float s = 0.f;
    for (int j = 0; j < N_NODES; ++j) {
      float e = fi + f2[j]; e = e > 0.f ? e : ALPHA * e;
      s += __expf(e - m);
    }
    rmx[tid] = m; rsi[tid] = 1.0f / s;
  }
  __syncthreads();

  // ---- hp = att @ h  (M=256, N=128, K=256); dual M-tiles share B-frags & f2 reads ----
  float* __restrict__ dst = gout + (size_t)u * BATCH * N_NODES * HID +
                            (size_t)b * HID * N_NODES;   // [j][n]
  {
    v8f acc0[8], acc1[8];
#pragma unroll
    for (int nt = 0; nt < 8; ++nt) {
      acc0[nt] = (v8f){0.f, 0.f, 0.f, 0.f, 0.f, 0.f, 0.f, 0.f};
      acc1[nt] = (v8f){0.f, 0.f, 0.f, 0.f, 0.f, 0.f, 0.f, 0.f};
    }
    const int m0 = mt0 * 16 + col, m1 = m0 + 16;
    const float fi0 = f1[m0], mx0 = rmx[m0], iv0 = rsi[m0];
    const float fi1 = f1[m1], mx1 = rmx[m1], iv1 = rsi[m1];
    for (int ks = 0; ks < 8; ++ks) {
      v16bf af0, af1;
#pragma unroll
      for (int e = 0; e < 16; ++e) {
        const int k = ks * 32 + ((e < 8 ? e : e + 8) + hi * 8);
        const float fk = f2[k];
        float e0 = fi0 + fk; e0 = e0 > 0.f ? e0 : ALPHA * e0;
        float e1 = fi1 + fk; e1 = e1 > 0.f ? e1 : ALPHA * e1;
        af0[e] = (bf16)(__expf(e0 - mx0) * iv0);
        af1[e] = (bf16)(__expf(e1 - mx1) * iv1);
      }
#pragma unroll
      for (int nt = 0; nt < 8; ++nt) {
        const v16bf bfr = load_frag_kmaj(sHt + nt * 16 * HLD + ks * 32, HLD);
        acc0[nt] = wmma_bf16(af0, bfr, acc0[nt]);
        acc1[nt] = wmma_bf16(af1, bfr, acc1[nt]);
      }
    }
    // elu + transposed store: (m,j) -> dst[j*256 + m]; 8 contiguous floats per lane
#pragma unroll
    for (int nt = 0; nt < 8; ++nt) {
      v4f o0, o1, o2, o3;
#pragma unroll
      for (int v = 0; v < 4; ++v) {
        float z;
        z = acc0[nt][v];     o0[v] = z > 0.f ? z : (__expf(z) - 1.0f);
        z = acc0[nt][v + 4]; o1[v] = z > 0.f ? z : (__expf(z) - 1.0f);
        z = acc1[nt][v];     o2[v] = z > 0.f ? z : (__expf(z) - 1.0f);
        z = acc1[nt][v + 4]; o3[v] = z > 0.f ? z : (__expf(z) - 1.0f);
      }
      float* p = dst + (size_t)(nt * 16 + col) * N_NODES + mt0 * 16 + hi * 8;
      *(v4f*)p = o0;
      *(v4f*)(p + 4) = o1;
      *(v4f*)(p + 16) = o2;
      *(v4f*)(p + 20) = o3;
    }
  }
}

// ---------------- LSTM gate combine: all buffers in (b, j, n) layout ----------------
__global__ __launch_bounds__(256) void combine_kernel(const float* __restrict__ gat,
                                                      const float* __restrict__ bi,
                                                      const float* __restrict__ bf_,
                                                      const float* __restrict__ bc,
                                                      const float* __restrict__ bo,
                                                      float* __restrict__ c,
                                                      float* __restrict__ hpre) {
  const size_t S = (size_t)BATCH * N_NODES * HID;
  const size_t i4 = (size_t)blockIdx.x * 256 + threadIdx.x;
  if (i4 >= S / 4) return;
  const size_t i = i4 * 4;
  const int j = (int)((i >> 8) & 127);
  const float vbi = bi[j], vbf = bf_[j], vbc = bc[j], vbo = bo[j];
  v4f g0 = *(const v4f*)(gat + 0 * S + i), g1 = *(const v4f*)(gat + 1 * S + i);
  v4f g2 = *(const v4f*)(gat + 2 * S + i), g3 = *(const v4f*)(gat + 3 * S + i);
  v4f g4 = *(const v4f*)(gat + 4 * S + i), g5 = *(const v4f*)(gat + 5 * S + i);
  v4f g6 = *(const v4f*)(gat + 6 * S + i), g7 = *(const v4f*)(gat + 7 * S + i);
  v4f cv = *(const v4f*)(c + i);
  v4f cn, hp;
#pragma unroll
  for (int v = 0; v < 4; ++v) {
    const float it = sigmf(g0[v] + g1[v] + vbi);
    const float ft = sigmf(g2[v] + g3[v] + vbf);
    const float gt = tanhf(g4[v] + g5[v] + vbc);
    const float ot = sigmf(g6[v] + g7[v] + vbo);
    cn[v] = ft * cv[v] + it * gt;
    hp[v] = ot * tanhf(cn[v]);
  }
  *(v4f*)(c + i) = cn;
  *(v4f*)(hpre + i) = hp;
}

// ---------------- f_att + residual + output write (one batch per WG) ----------------
__global__ __launch_bounds__(256) void fatt_kernel(const float* __restrict__ hpre,
                                                   const float* __restrict__ pool_w,
                                                   const float* __restrict__ pool_b,
                                                   const float* __restrict__ linh_w,
                                                   const float* __restrict__ linh_b,
                                                   const float* __restrict__ linq_w,
                                                   const float* __restrict__ linq_b,
                                                   const float* __restrict__ linu_w,
                                                   const float* __restrict__ linu_b,
                                                   float* __restrict__ hout,
                                                   float* __restrict__ seq_out, int t) {
  extern __shared__ char smem[];
  bf16*  sHP = (bf16*)smem;                  // hpre : [n][j], 256 x XLD (A operand)
  bf16*  sLW = sHP + N_NODES * XLD;          // linh_w rows = K-major B : [j][k], 128 x XLD
  float* sqv = (float*)(sLW + HID * XLD);    // 128: q
  float* sqj = sqv + HID;                    // 128: q@linq_w^T + linq_b + linh_b
  float* sa  = sqj + HID;                    // 256: pre-sigmoid attention scalar

  const int b = blockIdx.x, tid = threadIdx.x;
  const float* __restrict__ hp = hpre + (size_t)b * HID * N_NODES;  // [j][n]

  for (int i = tid; i < N_NODES * HID / 2; i += 256) {
    const int n = i & 255, j = (i >> 8) * 2;
    v2bf pk = {(bf16)hp[j * N_NODES + n], (bf16)hp[(j + 1) * N_NODES + n]};
    *(v2bf*)(sHP + n * XLD + j) = pk;
  }
  for (int i = tid; i < HID * HID / 4; i += 256) {
    const int j = i >> 5, k4 = (i & 31) << 2;
    const v4f wv4 = *(const v4f*)(linh_w + j * HID + k4);
    v4bf pk = {(bf16)wv4.x, (bf16)wv4.y, (bf16)wv4.z, (bf16)wv4.w};
    *(v4bf*)(sLW + j * XLD + k4) = pk;
  }
  sa[tid] = 0.f;
  __syncthreads();

  if (tid < HID) {
    float s = 0.f;
    for (int n = 0; n < N_NODES; ++n) s += pool_w[n] * hp[tid * N_NODES + n];
    sqv[tid] = fmaxf(s + pool_b[0], 0.f);   // relu
  }
  __syncthreads();
  if (tid < HID) {
    float s = 0.f;
    for (int k = 0; k < HID; ++k) s += sqv[k] * linq_w[tid * HID + k];
    sqj[tid] = s + linq_b[tid] + linh_b[tid];
  }
  __syncthreads();

  // t = tanh(hp @ linh_w^T + sqj); a_pre[n] = sum_j t[n,j]*linu_w[j]
  // Dual M-tiles per wave, shared B-frags; LDS float atomics for the j-reduction.
  const int wv = tid >> 5, lane = tid & 31, col = lane & 15, hi = lane >> 4;
  {
    v16bf a0[4], a1[4];
#pragma unroll
    for (int ks = 0; ks < 4; ++ks) {
      a0[ks] = load_frag_kmaj(sHP + (wv * 32) * XLD + ks * 32, XLD);
      a1[ks] = load_frag_kmaj(sHP + (wv * 32 + 16) * XLD + ks * 32, XLD);
    }
#pragma unroll
    for (int nt = 0; nt < 8; ++nt) {
      v8f c0 = {0.f, 0.f, 0.f, 0.f, 0.f, 0.f, 0.f, 0.f};
      v8f c1 = {0.f, 0.f, 0.f, 0.f, 0.f, 0.f, 0.f, 0.f};
#pragma unroll
      for (int ks = 0; ks < 4; ++ks) {
        const v16bf bfr = load_frag_kmaj(sLW + nt * 16 * XLD + ks * 32, XLD);
        c0 = wmma_bf16(a0[ks], bfr, c0);
        c1 = wmma_bf16(a1[ks], bfr, c1);
      }
      const int j = nt * 16 + col;
      const float lw = linu_w[j], qj = sqj[j];
#pragma unroll
      for (int v = 0; v < 8; ++v) {
        atomicAdd(&sa[wv * 32 + hi * 8 + v], tanhf(c0[v] + qj) * lw);
        atomicAdd(&sa[wv * 32 + 16 + hi * 8 + v], tanhf(c1[v] + qj) * lw);
      }
    }
  }
  __syncthreads();

  const size_t TB = (size_t)TSEQ * BATCH * N_NODES;  // 131072
  const float lub = linu_b[0];
  for (int i4 = tid; i4 < HID * N_NODES / 4; i4 += 256) {
    const int j = i4 >> 6, n4 = (i4 & 63) << 2;
    const v4f hv = *(const v4f*)(hp + (size_t)j * N_NODES + n4);
    v4f hn;
#pragma unroll
    for (int v = 0; v < 4; ++v) {
      const float a = sigmf(sa[n4 + v] + lub);
      hn[v] = hv[v] * (1.0f + a);                    // f_att(h) + h
    }
    *(v4f*)(hout + (size_t)b * HID * N_NODES + (size_t)j * N_NODES + n4) = hn;
    *(v4f*)(seq_out + (size_t)j * TB + (size_t)t * (BATCH * N_NODES) + b * N_NODES + n4) = hn;
  }
}

// (b, j, n) -> reference (b*256+n, j) row-major; iteration is write-coalesced.
__global__ __launch_bounds__(256) void transpose_out_kernel(const float* __restrict__ src,
                                                            float* __restrict__ dst) {
  const size_t S = (size_t)BATCH * N_NODES * HID;
  const size_t i = (size_t)blockIdx.x * 256 + threadIdx.x;
  if (i >= S) return;
  const int j = (int)(i & 127);
  const size_t bn = i >> 7;
  const int n = (int)(bn & 255);
  const size_t b = bn >> 8;
  dst[i] = src[b * (size_t)(HID * N_NODES) + (size_t)j * N_NODES + n];
}

__global__ __launch_bounds__(256) void zero_kernel(float* __restrict__ p, size_t n) {
  const size_t i = (size_t)blockIdx.x * 256 + threadIdx.x;
  if (i < n) p[i] = 0.f;
}

extern "C" void kernel_launch(void* const* d_in, const int* in_sizes, int n_in,
                              void* d_out, int out_size, void* d_ws, size_t ws_size,
                              hipStream_t stream) {
  const float* x = (const float*)d_in[0];
  GatArgs ga;
  const float* bg[4];
  for (int g = 0; g < 4; ++g) {   // dict order per gate: W_W, W_a, U_W, U_a, b
    ga.W[g * 2 + 0] = (const float*)d_in[1 + g * 5 + 0];
    ga.a[g * 2 + 0] = (const float*)d_in[1 + g * 5 + 1];
    ga.W[g * 2 + 1] = (const float*)d_in[1 + g * 5 + 2];
    ga.a[g * 2 + 1] = (const float*)d_in[1 + g * 5 + 3];
    bg[g]           = (const float*)d_in[1 + g * 5 + 4];
  }
  const float* pool_w = (const float*)d_in[21];
  const float* pool_b = (const float*)d_in[22];
  const float* linh_w = (const float*)d_in[23];
  const float* linh_b = (const float*)d_in[24];
  const float* linq_w = (const float*)d_in[25];
  const float* linq_b = (const float*)d_in[26];
  const float* linu_w = (const float*)d_in[27];
  const float* linu_b = (const float*)d_in[28];

  const size_t S = (size_t)BATCH * N_NODES * HID;   // 1,048,576 per state buffer
  float* ws     = (float*)d_ws;
  float* gatbuf = ws;                                // 8*S, layout [u][b][j][n]
  float* h_cur  = ws + 8 * S;                        // [b][j][n]
  float* c_cur  = h_cur + S;                         // [b][j][n]
  float* hpre   = c_cur + S;                         // [b][j][n]
  float* out    = (float*)d_out;

  const size_t GAT_LDS  = (size_t)(HID * XLD + N_NODES * XLD + HID * HLD) * sizeof(bf16) +
                          4 * N_NODES * sizeof(float);                 // 176,128 B
  const size_t FATT_LDS = (size_t)(N_NODES * XLD + HID * XLD) * sizeof(bf16) +
                          (2 * HID + N_NODES) * sizeof(float);         // 106,496 B
  hipFuncSetAttribute((const void*)gat_kernel,
                      hipFuncAttributeMaxDynamicSharedMemorySize, (int)GAT_LDS);
  hipFuncSetAttribute((const void*)fatt_kernel,
                      hipFuncAttributeMaxDynamicSharedMemorySize, (int)FATT_LDS);

  zero_kernel<<<(unsigned)((2 * S + 255) / 256), 256, 0, stream>>>(h_cur, 2 * S);

  for (int t = 0; t < TSEQ; ++t) {
    gat_kernel<<<dim3(BATCH, 8), 256, GAT_LDS, stream>>>(x, h_cur, ga, gatbuf, t);
    combine_kernel<<<(unsigned)((S / 4 + 255) / 256), 256, 0, stream>>>(
        gatbuf, bg[0], bg[1], bg[2], bg[3], c_cur, hpre);
    fatt_kernel<<<BATCH, 256, FATT_LDS, stream>>>(hpre, pool_w, pool_b, linh_w, linh_b,
                                                  linq_w, linq_b, linu_w, linu_b,
                                                  h_cur, out, t);
  }

  const size_t SEQ = (size_t)HID * TSEQ * BATCH * N_NODES;  // 16,777,216
  transpose_out_kernel<<<(unsigned)((S + 255) / 256), 256, 0, stream>>>(h_cur, out + SEQ);
  transpose_out_kernel<<<(unsigned)((S + 255) / 256), 256, 0, stream>>>(c_cur, out + SEQ + S);
}
